// RBFNet_79242146611354
// MI455X (gfx1250) — matrix operand order repaired and checked by reading, not compiled, not run on hardware
//
#include <hip/hip_runtime.h>
#include <hip/hip_bf16.h>

typedef __attribute__((ext_vector_type(16))) _Float16 v16h;
typedef __attribute__((ext_vector_type(8)))  float    v8f;

#define LOG2E 1.4426950408889634f

// cross-lane add with xor partner via ds_swizzle (group-of-32, xor mask in bits [14:10])
template <int OFF>
__device__ __forceinline__ float swz_add(float v) {
  return v + __int_as_float(__builtin_amdgcn_ds_swizzle(__float_as_int(v), OFF));
}

__device__ __forceinline__ float lane_bcast(float v, int srcLane) {
  return __int_as_float(__builtin_amdgcn_ds_bpermute(srcLane << 2, __float_as_int(v)));
}

__device__ __forceinline__ v16h pack16(float4 a, float4 b, float4 c, float4 d) {
  v16h r;
  r[0]=(_Float16)a.x;  r[1]=(_Float16)a.y;  r[2]=(_Float16)a.z;  r[3]=(_Float16)a.w;
  r[4]=(_Float16)b.x;  r[5]=(_Float16)b.y;  r[6]=(_Float16)b.z;  r[7]=(_Float16)b.w;
  r[8]=(_Float16)c.x;  r[9]=(_Float16)c.y;  r[10]=(_Float16)c.z; r[11]=(_Float16)c.w;
  r[12]=(_Float16)d.x; r[13]=(_Float16)d.y; r[14]=(_Float16)d.z; r[15]=(_Float16)d.w;
  return r;
}

__device__ __forceinline__ float sq4(float4 a) {
  return a.x*a.x + a.y*a.y + a.z*a.z + a.w*a.w;
}

// y[b] = sigmoid( sum_j exp( t_j*(2*dot(x_b,mu_j) - |x_b|^2 - |mu_j|^2) ) * lin_j + bias ),
// t_j = 0.5/sig_j^2.  GEMM via v_wmma_f32_16x16x32_f16 (K=64 in 2 steps, N=64 in 4 tiles).
// All exponentials in base-2 with log2(e) pre-folded into the per-center constants:
//   res = exp2( t2'_j*(dot - 0.5*|x|^2) - a'_j ),  t2'_j = 2*t_j*log2e,  a'_j = t_j*|mu_j|^2*log2e.
__global__ __launch_bounds__(256) void rbf_wmma_kernel(
    const float* __restrict__ x, const float* __restrict__ mus,
    const float* __restrict__ sigs, const float* __restrict__ lin,
    const float* __restrict__ bias, float* __restrict__ y, int ntiles) {
  const int lane = threadIdx.x & 31;
  const int n    = lane & 15;   // column (B/C) or row (A) index within 16
  const int h    = lane >> 4;   // half-wave selector
  const int wid  = blockIdx.x * (blockDim.x >> 5) + (threadIdx.x >> 5);
  const int nwaves = gridDim.x * (blockDim.x >> 5);

  // ---------- per-wave prep: B matrix (all 64 centers) in registers + per-center constants ----------
  // B layout (32x16 f16, contiguous k order chosen here, not the ISA table order): lane n = column, lanes 0-15 hold K=0..15,
  // lanes 16-31 hold K=16..31, packed 2 halfs per VGPR in K order.
  v16h Breg[8];                       // [ct*2 + kstep]
  float t2[4], aj[4], lj[4];
#pragma unroll
  for (int ct = 0; ct < 4; ++ct) {
    const int j = ct * 16 + n;
    const float* mr = mus + j * 64;
    float msq = 0.f;
#pragma unroll
    for (int kk = 0; kk < 2; ++kk) {
      const float* p0 = mr + kk * 32 + 16 * h;   // 16 consecutive k values, 16B aligned
      float4 g0 = ((const float4*)p0)[0];
      float4 g1 = ((const float4*)p0)[1];
      float4 g2 = ((const float4*)p0)[2];
      float4 g3 = ((const float4*)p0)[3];
      Breg[ct * 2 + kk] = pack16(g0, g1, g2, g3);
      msq += sq4(g0) + sq4(g1) + sq4(g2) + sq4(g3);
    }
    msq = swz_add<0x401F>(msq);       // xor-16: combine K halves -> |mu_j|^2
    const float sg = sigs[j];
    const float t  = 0.5f / (sg * sg);
    t2[ct] = (t + t) * LOG2E;
    aj[ct] = t * msq * LOG2E;
    lj[ct] = lin[j];
  }
  const float sig_c = -bias[0] * LOG2E;   // sigmoid: 1/(1 + exp2(-log2e*s - log2e*bias))

  // ---------- grid-stride loop over 16-row tiles of x ----------
  for (int t = wid; t < ntiles; t += nwaves) {
    const float* xr = x + ((size_t)t * 16 + n) * 64;   // A row = lane&15
    if (t + nwaves < ntiles)
      __builtin_prefetch(x + ((size_t)(t + nwaves) * 16 + n) * 64, 0, 0);

    v8f acc[4] = {};                  // 4 column tiles of 16x16 f32 accumulators
    float xs = 0.f;                   // per-lane partial |x_row|^2 (f32, exact)
#pragma unroll
    for (int kk = 0; kk < 2; ++kk) {
      const int kb = kk * 32;
      // A layout (16x32 f16): halfs 0..15 -> K = kb+16h+0..15 (matches B).
      // Two aligned 32B chunks per lane.
      const float* pa = xr + kb + 16 * h;
      float4 f0 = ((const float4*)pa)[0];
      float4 f1 = ((const float4*)pa)[1];
      const float* pb = xr + kb + 8 + 16 * h;
      float4 f2 = ((const float4*)pb)[0];
      float4 f3 = ((const float4*)pb)[1];
      v16h a = pack16(f0, f1, f2, f3);
      xs += sq4(f0) + sq4(f1) + sq4(f2) + sq4(f3);
#pragma unroll
      for (int ct = 0; ct < 4; ++ct) {
        acc[ct] = __builtin_amdgcn_wmma_f32_16x16x32_f16(
            false, a, false, Breg[ct * 2 + kk], (short)0, acc[ct], false, false);
      }
    }
    xs = swz_add<0x401F>(xs);         // all lanes: |x|^2 for row (lane&15)

    // C/D layout: element v of acc -> row M = v + 8h, column N = lane&15.
    float hx[8];                      // 0.5*|x_row|^2 for the 8 rows this lane accumulates
#pragma unroll
    for (int v = 0; v < 8; ++v) hx[v] = 0.5f * lane_bcast(xs, 8 * h + v);

    float p[8] = {};
#pragma unroll
    for (int ct = 0; ct < 4; ++ct) {
#pragma unroll
      for (int v = 0; v < 8; ++v) {
        const float d   = acc[ct][v] - hx[v];                     // dot - 0.5*|x|^2
        const float e   = __builtin_amdgcn_exp2f(fmaf(t2[ct], d, -aj[ct]));
        p[v] = fmaf(e, lj[ct], p[v]);                             // RBF activation * lin_j
      }
    }

    // butterfly-reduce each row's partial over the 16 lanes of this half-wave (LDS pipe)
#pragma unroll
    for (int v = 0; v < 8; ++v) {
      float s = p[v];
      s = swz_add<0x041F>(s);   // xor 1
      s = swz_add<0x081F>(s);   // xor 2
      s = swz_add<0x101F>(s);   // xor 4
      s = swz_add<0x201F>(s);   // xor 8
      // sigmoid(s + bias) = rcp(1 + exp2(-log2e*(s + bias)))
      const float w = __builtin_amdgcn_exp2f(fmaf(s, -LOG2E, sig_c));
      p[v] = __builtin_amdgcn_rcpf(1.0f + w);
    }

    if (n == 0) {                                  // lanes 0 and 16 write 8 rows each
      float* yo = y + (size_t)t * 16 + 8 * h;
      ((float4*)yo)[0] = make_float4(p[0], p[1], p[2], p[3]);
      ((float4*)yo)[1] = make_float4(p[4], p[5], p[6], p[7]);
    }
  }
}

extern "C" void kernel_launch(void* const* d_in, const int* in_sizes, int n_in,
                              void* d_out, int out_size, void* d_ws, size_t ws_size,
                              hipStream_t stream) {
  (void)n_in; (void)d_ws; (void)ws_size; (void)in_sizes;
  const float* x    = (const float*)d_in[0];
  const float* mus  = (const float*)d_in[1];
  const float* sigs = (const float*)d_in[2];
  const float* lin  = (const float*)d_in[3];
  const float* bias = (const float*)d_in[4];
  float* y = (float*)d_out;

  const int rows   = out_size;        // B (2,000,000), divisible by 16
  const int ntiles = rows / 16;
  const int blocks = 1024;            // 8192 waves; ~15 tiles/wave amortizes B-reg prep
  rbf_wmma_kernel<<<blocks, 256, 0, stream>>>(x, mus, sigs, lin, bias, y, ntiles);
}
